// NMultiheadSelfAttention_10651518894456
// MI455X (gfx1250) — matrix operand order repaired and checked by reading, not compiled, not run on hardware
//
#include <hip/hip_runtime.h>
#include <hip/hip_bf16.h>

// ---------------------------------------------------------------------------
// Problem constants (B=4, S=1024, E=1024, H=16, D=64)
// ---------------------------------------------------------------------------
#define BSZ   4
#define SSEQ  1024
#define EMB   1024
#define HH_N  16
#define DH    64
#define NTOK  (BSZ * SSEQ)          // 4096 rows
#define SCALE_INV 0.125f            // 1/sqrt(64)

typedef __bf16 bf16_t;
typedef __attribute__((ext_vector_type(16))) bf16_t v16bf;
typedef __attribute__((ext_vector_type(8)))  bf16_t v8bf;
typedef __attribute__((ext_vector_type(4)))  bf16_t v4bf;
typedef __attribute__((ext_vector_type(8)))  float  v8f;
typedef __attribute__((ext_vector_type(4)))  float  v4f;
typedef __attribute__((ext_vector_type(4)))  int    v4i;

union ABFrag { v16bf full; v8bf half[2]; };

// ---------------------------------------------------------------------------
// CDNA5 async global->LDS copy (ASYNCcnt-tracked), with safe fallback
// Builtin signature (probe-confirmed by round-2 diagnostic):
//   (v4i addrspace(1)* src, v4i addrspace(3)* dst, imm int offset, imm int cpol)
// ---------------------------------------------------------------------------
#if defined(__has_builtin)
#if __has_builtin(__builtin_amdgcn_global_load_async_to_lds_b128)
#define HAVE_ASYNC_LDS 1
#endif
#if __has_builtin(__builtin_amdgcn_s_wait_asynccnt)
#define HAVE_WAIT_ASYNC_BUILTIN 1
#endif
#endif
#ifndef HAVE_ASYNC_LDS
#define HAVE_ASYNC_LDS 0
#endif

#if HAVE_ASYNC_LDS
typedef __attribute__((address_space(1))) v4i* gv4i_p;
typedef __attribute__((address_space(3))) v4i* lv4i_p;
__device__ __forceinline__ void async_cp16(const bf16_t* g, bf16_t* l) {
  __builtin_amdgcn_global_load_async_to_lds_b128(
      (gv4i_p)(void*)g, (lv4i_p)(void*)l, /*offset=*/0, /*cpol=*/0);
}
#if defined(HAVE_WAIT_ASYNC_BUILTIN)
#define WAIT_ASYNC(n) __builtin_amdgcn_s_wait_asynccnt(n)
#else
#define WAIT_ASYNC(n) asm volatile("s_wait_asynccnt %0" ::"i"(n))
#endif
#else
#define WAIT_ASYNC(n)
#endif

// ---------------------------------------------------------------------------
// f32 -> bf16 elementwise conversion (vectorized x4)
// ---------------------------------------------------------------------------
__global__ __launch_bounds__(256) void cvt_f32_bf16_kernel(
    const float* __restrict__ in, bf16_t* __restrict__ out, long n) {
  long i = ((long)blockIdx.x * blockDim.x + threadIdx.x) * 4;
  if (i + 3 < n) {
    v4f f = *(const v4f*)(in + i);
    v4bf o;
    o[0] = (bf16_t)f[0]; o[1] = (bf16_t)f[1];
    o[2] = (bf16_t)f[2]; o[3] = (bf16_t)f[3];
    *(v4bf*)(out + i) = o;
  }
}

// ---------------------------------------------------------------------------
// t[b,h,d] = sum_s p2[b,s,h*64+d]   (grid = B*H blocks, 256 threads)
// ---------------------------------------------------------------------------
__global__ __launch_bounds__(256) void colsum_kernel(
    const bf16_t* __restrict__ p2, float* __restrict__ t) {
  int z = blockIdx.x;                  // b*H + h
  int b = z >> 4, h = z & 15;
  int d = threadIdx.x & 63;
  int g = threadIdx.x >> 6;            // 0..3
  const bf16_t* base = p2 + (long)b * SSEQ * EMB + h * DH + d;
  float acc = 0.f;
  for (int s = g; s < SSEQ; s += 4) acc += (float)base[(long)s * EMB];
  __shared__ float red[256];
  red[threadIdx.x] = acc;
  __syncthreads();
  if (g == 0)
    t[z * DH + d] = red[d] + red[64 + d] + red[128 + d] + red[192 + d];
}

// ---------------------------------------------------------------------------
// Mt[z][c][a] = (1/8) * sum_d A[a,c,d] * t[z,d]   (bf16 out, pre-transposed)
// ---------------------------------------------------------------------------
__global__ __launch_bounds__(256) void buildM_kernel(
    const float* __restrict__ A, const float* __restrict__ t,
    bf16_t* __restrict__ Mt) {
  int z = blockIdx.x;
  __shared__ float ts[64];
  if (threadIdx.x < 64) ts[threadIdx.x] = t[z * 64 + threadIdx.x];
  __syncthreads();
  for (int idx = threadIdx.x; idx < 4096; idx += blockDim.x) {
    int c = idx >> 6, a = idx & 63;
    const float* ap = A + ((long)a * 64 + c) * 64;
    float acc = 0.f;
#pragma unroll 8
    for (int d = 0; d < 64; ++d) acc += ap[d] * ts[d];
    Mt[(long)z * 4096 + idx] = (bf16_t)(acc * SCALE_INV);
  }
}

// ---------------------------------------------------------------------------
// Row softmax in place, rows of length S=1024, one block (256 thr) per row
// ---------------------------------------------------------------------------
__global__ __launch_bounds__(256) void softmax_kernel(float* __restrict__ attn) {
  long row = blockIdx.x;
  float* p = attn + row * SSEQ;
  int tid = threadIdx.x;
  v4f v = *(v4f*)(p + tid * 4);
  __shared__ float red[256];
  float m = fmaxf(fmaxf(v[0], v[1]), fmaxf(v[2], v[3]));
  red[tid] = m;
  __syncthreads();
  for (int s = 128; s > 0; s >>= 1) {
    if (tid < s) red[tid] = fmaxf(red[tid], red[tid + s]);
    __syncthreads();
  }
  m = red[0];
  __syncthreads();
  v[0] = __expf(v[0] - m); v[1] = __expf(v[1] - m);
  v[2] = __expf(v[2] - m); v[3] = __expf(v[3] - m);
  red[tid] = v[0] + v[1] + v[2] + v[3];
  __syncthreads();
  for (int s = 128; s > 0; s >>= 1) {
    if (tid < s) red[tid] += red[tid + s];
    __syncthreads();
  }
  float inv = 1.0f / red[0];
  v[0] *= inv; v[1] *= inv; v[2] *= inv; v[3] *= inv;
  *(v4f*)(p + tid * 4) = v;
}

// ---------------------------------------------------------------------------
// vT[b,h,d,s] = v[b,s,h*64+d]  (per-head 64x64 tile transpose)
// grid = (B*H, S/64), 256 threads
// ---------------------------------------------------------------------------
__global__ __launch_bounds__(256) void transpose_head_kernel(
    const bf16_t* __restrict__ vsrc, bf16_t* __restrict__ vT) {
  int z = blockIdx.x;                  // b*H + h
  int b = z >> 4, h = z & 15;
  int s0 = blockIdx.y * 64;
  __shared__ bf16_t tile[64][72];
  int sl = threadIdx.x >> 2;           // 0..63
  int ch = (threadIdx.x & 3) * 16;     // 0/16/32/48
  const bf16_t* src = vsrc + ((long)b * SSEQ + s0 + sl) * EMB + h * DH + ch;
  *(v8bf*)&tile[sl][ch]     = *(const v8bf*)src;
  *(v8bf*)&tile[sl][ch + 8] = *(const v8bf*)(src + 8);
  __syncthreads();
  int dl = sl;                         // destination d-index
  bf16_t ov[16];
#pragma unroll
  for (int k = 0; k < 16; ++k) ov[k] = tile[ch + k][dl];
  bf16_t* dst = vT + ((long)z * DH + dl) * SSEQ + s0 + ch;
  *(v8bf*)dst       = *(v8bf*)&ov[0];
  *(v8bf*)(dst + 8) = *(v8bf*)&ov[8];
}

// ---------------------------------------------------------------------------
// Batched WMMA GEMM:  C[m,n] = sum_k A[m,k] * B[n,k]  (+ bias[n])
// Block: 256 thr (8 wave32), tile 128x128, K-step 32, wave tile 32x64.
// bf16-A path: double-buffered GLOBAL_LOAD_ASYNC_TO_LDS_B128 staging.
// f32-A path : register staging with f32->bf16 conversion.
// OOB rows are clamped (not zero-filled): every M here is a multiple of 128,
// and B-tile columns >= N only feed accumulators that are never stored.
// ---------------------------------------------------------------------------
#define BM 128
#define BN 128
#define BK 32
#define LDSW 40     // padded LDS row stride (elements)

__device__ __forceinline__ v8bf load8(const bf16_t* p) {
  return *(const v8bf*)p;
}
__device__ __forceinline__ v8bf load8(const float* p) {
  v4f a = *(const v4f*)p;
  v4f b = *(const v4f*)(p + 4);
  v8bf r;
#pragma unroll
  for (int i = 0; i < 4; ++i) { r[i] = (bf16_t)a[i]; r[i + 4] = (bf16_t)b[i]; }
  return r;
}

template <typename TA>
__global__ __launch_bounds__(256) void wmma_gemm_kernel(
    const TA* __restrict__ Abase, const bf16_t* __restrict__ Bbase,
    const float* __restrict__ bias,
    float* __restrict__ Cf, bf16_t* __restrict__ Cb,
    int Mrows, int Ncols, int Kdim,
    long lda, long ldb, long ldcf, long ldcb,
    int HH,
    long sA0, long sA1, long sB0, long sB1,
    long sCf0, long sCf1, long sCb0, long sCb1) {
  constexpr bool kAsync = (HAVE_ASYNC_LDS != 0) && (sizeof(TA) == 2);

  __shared__ bf16_t sA[2][BM * LDSW];
  __shared__ bf16_t sB[2][BN * LDSW];

  int z = blockIdx.z;
  int zq = z / HH, zr = z % HH;
  const TA*     Ap = Abase + zq * sA0 + zr * sA1;
  const bf16_t* Bp = Bbase + zq * sB0 + zr * sB1;

  int m0 = blockIdx.y * BM;
  int n0 = blockIdx.x * BN;

  int tid  = threadIdx.x;
  int lane = tid & 31;
  int wid  = tid >> 5;
  int wm   = wid & 3;           // wave M offset = wm*32
  int wn   = wid >> 2;          // wave N offset = wn*64

  v8f acc[2][4];
#pragma unroll
  for (int i = 0; i < 2; ++i)
#pragma unroll
    for (int j = 0; j < 4; ++j)
#pragma unroll
      for (int r = 0; r < 8; ++r) acc[i][j][r] = 0.f;

  // staging geometry: each thread owns 2 A-chunks + 2 B-chunks of 8 elements
  int r0 = tid >> 2;            // 0..63
  int r1 = r0 + 64;             // 64..127
  int cc = (tid & 3) * 8;       // K chunk within 32

  int am0 = min(m0 + r0, Mrows - 1);
  int am1 = min(m0 + r1, Mrows - 1);
  int bn0 = min(n0 + r0, Ncols - 1);
  int bn1 = min(n0 + r1, Ncols - 1);

  const TA*     aP0 = Ap + (long)am0 * lda + cc;
  const TA*     aP1 = Ap + (long)am1 * lda + cc;
  const bf16_t* bP0 = Bp + (long)bn0 * ldb + cc;
  const bf16_t* bP1 = Bp + (long)bn1 * ldb + cc;

  unsigned la0 = (unsigned)(r0 * LDSW + cc);
  unsigned la1 = (unsigned)(r1 * LDSW + cc);

  auto stage = [&](int buf, int k0) {
    if constexpr (kAsync) {
#if HAVE_ASYNC_LDS
      async_cp16((const bf16_t*)(aP0 + k0), &sA[buf][la0]);
      async_cp16((const bf16_t*)(aP1 + k0), &sA[buf][la1]);
      async_cp16(bP0 + k0, &sB[buf][la0]);
      async_cp16(bP1 + k0, &sB[buf][la1]);
#endif
    } else {
      *(v8bf*)&sA[buf][la0] = load8(aP0 + k0);
      *(v8bf*)&sA[buf][la1] = load8(aP1 + k0);
      *(v8bf*)&sB[buf][la0] = load8(bP0 + k0);
      *(v8bf*)&sB[buf][la1] = load8(bP1 + k0);
    }
  };

  // ISA fragment addressing (16-bit operands, wave32):
  //  A 16x32: lanes 0-15 row M hold K 0-7 & 16-23; lanes 16-31 hold 8-15 & 24-31
  //  B 32x16: lanes 0-15 col N hold K 0-7 & 16-23; lanes 16-31 hold 8-15 & 24-31
  int fm    = lane & 15;
  int kgrpA = (lane >> 4) * 8;
  int kgrpB = (lane >> 4) * 8;

  int nk = Kdim / BK;
  stage(0, 0);

  for (int k = 0; k < nk; ++k) {
    int cur = k & 1;
    if (k + 1 < nk) stage(1 - cur, (k + 1) * BK);

    if constexpr (kAsync) {
      // async ops complete in order: <=4 outstanding means tile k is resident
      if (k + 1 < nk) { WAIT_ASYNC(4); } else { WAIT_ASYNC(0); }
    }
    __syncthreads();

    ABFrag af[2], bfr[4];
#pragma unroll
    for (int i = 0; i < 2; ++i) {
      const bf16_t* base = &sA[cur][(wm * 32 + i * 16 + fm) * LDSW + kgrpA];
      af[i].half[0] = *(const v8bf*)base;
      af[i].half[1] = *(const v8bf*)(base + 16);
    }
#pragma unroll
    for (int j = 0; j < 4; ++j) {
      const bf16_t* base = &sB[cur][(wn * 64 + j * 16 + fm) * LDSW + kgrpB];
      bfr[j].half[0] = *(const v8bf*)base;
      bfr[j].half[1] = *(const v8bf*)(base + 16);
    }

#pragma unroll
    for (int i = 0; i < 2; ++i)
#pragma unroll
      for (int j = 0; j < 4; ++j)
        acc[i][j] = __builtin_amdgcn_wmma_f32_16x16x32_bf16(
            false, af[i].full, false, bfr[j].full,
            (short)0, acc[i][j], false, false);

    __syncthreads();
  }

  // ---- epilogue: C/D layout — VGPR r: lanes 0-15 -> M=r, lanes 16-31 -> M=8+r
  int  gcol0 = n0 + wn * 64 + (lane & 15);
  int  grow0 = m0 + wm * 32 + (lane >> 4) * 8;
  long cfo = Cf ? (zq * sCf0 + zr * sCf1) : 0;
  long cbo = Cb ? (zq * sCb0 + zr * sCb1) : 0;

#pragma unroll
  for (int i = 0; i < 2; ++i)
#pragma unroll
    for (int j = 0; j < 4; ++j) {
      int gcol = gcol0 + j * 16;
      float bv = (bias && gcol < Ncols) ? bias[gcol] : 0.f;
#pragma unroll
      for (int r = 0; r < 8; ++r) {
        int grow = grow0 + i * 16 + r;
        if (grow < Mrows && gcol < Ncols) {
          float val = acc[i][j][r] + bv;
          if (Cf) Cf[cfo + (long)grow * ldcf + gcol] = val;
          if (Cb) Cb[cbo + (long)grow * ldcb + gcol] = (bf16_t)val;
        }
      }
    }
}

// ---------------------------------------------------------------------------
// Host-side launch
// ---------------------------------------------------------------------------
template <typename TA>
static void launch_gemm(hipStream_t s, const TA* A, const bf16_t* B,
                        const float* bias, float* Cf, bf16_t* Cb,
                        int M, int N, int K,
                        long lda, long ldb, long ldcf, long ldcb,
                        int batch, int HH,
                        long sA0, long sA1, long sB0, long sB1,
                        long sCf0, long sCf1, long sCb0, long sCb1) {
  dim3 grid((N + BN - 1) / BN, (M + BM - 1) / BM, batch);
  wmma_gemm_kernel<TA><<<grid, 256, 0, s>>>(
      A, B, bias, Cf, Cb, M, N, K, lda, ldb, ldcf, ldcb,
      HH, sA0, sA1, sB0, sB1, sCf0, sCf1, sCb0, sCb1);
}

static void launch_cvt(hipStream_t s, const float* in, bf16_t* out, long n) {
  long nthr = n / 4;
  cvt_f32_bf16_kernel<<<(unsigned)((nthr + 255) / 256), 256, 0, s>>>(in, out, n);
}

extern "C" void kernel_launch(void* const* d_in, const int* in_sizes, int n_in,
                              void* d_out, int out_size, void* d_ws, size_t ws_size,
                              hipStream_t stream) {
  (void)in_sizes; (void)n_in; (void)out_size; (void)ws_size;

  const float* x  = (const float*)d_in[0];
  const float* W0 = (const float*)d_in[1];
  const float* b0 = (const float*)d_in[2];
  const float* W1 = (const float*)d_in[3];
  const float* b1 = (const float*)d_in[4];
  const float* W2 = (const float*)d_in[5];
  const float* b2 = (const float*)d_in[6];
  const float* Wv = (const float*)d_in[7];
  const float* bv = (const float*)d_in[8];
  const float* Wo = (const float*)d_in[9];
  const float* bo = (const float*)d_in[10];
  const float* At = (const float*)d_in[11];   // (D,D,D) core tensor

  float* out  = (float*)d_out;                          // (B,S,E)
  float* attn = out + (size_t)NTOK * EMB;               // (B,H,S,S)

  // ---- carve workspace (~75 MB) ----
  size_t off = 0;
  auto carve = [&](size_t bytes) -> void* {
    void* p = (char*)d_ws + off;
    off += (bytes + 255) & ~(size_t)255;
    return p;
  };
  const size_t NE = (size_t)NTOK * EMB;   // 4M elements
  const size_t WE = (size_t)EMB * EMB;    // 1M elements
  bf16_t* xb  = (bf16_t*)carve(NE * 2);
  bf16_t* w0b = (bf16_t*)carve(WE * 2);
  bf16_t* w1b = (bf16_t*)carve(WE * 2);
  bf16_t* w2b = (bf16_t*)carve(WE * 2);
  bf16_t* wvb = (bf16_t*)carve(WE * 2);
  bf16_t* wob = (bf16_t*)carve(WE * 2);
  bf16_t* p0  = (bf16_t*)carve(NE * 2);
  bf16_t* p1  = (bf16_t*)carve(NE * 2);
  bf16_t* p2  = (bf16_t*)carve(NE * 2);
  bf16_t* vb  = (bf16_t*)carve(NE * 2);
  float*  tsum = (float*)carve((size_t)BSZ * HH_N * DH * 4);
  bf16_t* Mt  = (bf16_t*)carve((size_t)BSZ * HH_N * DH * DH * 2);
  bf16_t* q0t = (bf16_t*)carve(NE * 2);
  bf16_t* vT  = (bf16_t*)carve((size_t)BSZ * HH_N * DH * SSEQ * 2);
  bf16_t* vals = (bf16_t*)carve(NE * 2);

  const long SE = (long)SSEQ * EMB;       // per-batch stride in (B,S,E)
  const long SS = (long)SSEQ * SSEQ;      // per-head attn stride

  // 1) convert inputs to bf16
  launch_cvt(stream, x,  xb,  (long)NE);
  launch_cvt(stream, W0, w0b, (long)WE);
  launch_cvt(stream, W1, w1b, (long)WE);
  launch_cvt(stream, W2, w2b, (long)WE);
  launch_cvt(stream, Wv, wvb, (long)WE);
  launch_cvt(stream, Wo, wob, (long)WE);

  // 2) projections: p = x @ W^T + b   (4096 x 1024 x 1024)
  launch_gemm<bf16_t>(stream, xb, w0b, b0, nullptr, p0, NTOK, EMB, EMB,
                      EMB, EMB, 0, EMB, 1, 1, 0,0, 0,0, 0,0, 0,0);
  launch_gemm<bf16_t>(stream, xb, w1b, b1, nullptr, p1, NTOK, EMB, EMB,
                      EMB, EMB, 0, EMB, 1, 1, 0,0, 0,0, 0,0, 0,0);
  launch_gemm<bf16_t>(stream, xb, w2b, b2, nullptr, p2, NTOK, EMB, EMB,
                      EMB, EMB, 0, EMB, 1, 1, 0,0, 0,0, 0,0, 0,0);
  launch_gemm<bf16_t>(stream, xb, wvb, bv, nullptr, vb, NTOK, EMB, EMB,
                      EMB, EMB, 0, EMB, 1, 1, 0,0, 0,0, 0,0, 0,0);

  // 3) t[b,h,d] = sum_s p2 ;  Mt[z][c][a] = (1/8) sum_d A[a,c,d] t[z,d]
  colsum_kernel<<<BSZ * HH_N, 256, 0, stream>>>(p2, tsum);
  buildM_kernel<<<BSZ * HH_N, 256, 0, stream>>>(At, tsum, Mt);

  // 4) q0t = p0 @ Mt^T  per head (S x 64 x 64, 64 batches)
  launch_gemm<bf16_t>(stream, p0, Mt, nullptr, nullptr, q0t,
                      SSEQ, DH, DH, EMB, DH, 0, EMB,
                      BSZ * HH_N, HH_N,
                      SE, DH, (long)HH_N * DH * DH, (long)DH * DH,
                      0, 0, SE, DH);

  // 5) logits = q0t @ p1^T per head -> attn region (f32, already scaled)
  launch_gemm<bf16_t>(stream, q0t, p1, nullptr, attn, nullptr,
                      SSEQ, SSEQ, DH, EMB, EMB, SSEQ, 0,
                      BSZ * HH_N, HH_N,
                      SE, DH, SE, DH,
                      (long)HH_N * SS, SS, 0, 0);

  // 6) softmax rows in place (B*H*S rows of length S)
  softmax_kernel<<<BSZ * HH_N * SSEQ, 256, 0, stream>>>(attn);

  // 7) vT[b,h,d,s] = v[b,s,h*64+d]
  transpose_head_kernel<<<dim3(BSZ * HH_N, SSEQ / 64), 256, 0, stream>>>(vb, vT);

  // 8) values = attn(f32) @ vT^T per head -> packed back into (B,S,E) bf16
  launch_gemm<float>(stream, attn, vT, nullptr, nullptr, vals,
                     SSEQ, DH, SSEQ, SSEQ, SSEQ, 0, EMB,
                     BSZ * HH_N, HH_N,
                     (long)HH_N * SS, SS, (long)HH_N * DH * SSEQ, (long)DH * SSEQ,
                     0, 0, SE, DH);

  // 9) out = values @ Wo^T + bo  (f32 to d_out)
  launch_gemm<bf16_t>(stream, vals, wob, bo, out, nullptr,
                      NTOK, EMB, EMB, EMB, EMB, EMB, 0,
                      1, 1, 0,0, 0,0, 0,0, 0,0);
}
